// MaskDecoderLayer_20040317403251
// MI455X (gfx1250) — compile-verified, matrix operand order checked by reading, not run on hardware
//
#include <hip/hip_runtime.h>
#include <hip/hip_bf16.h>
#include <math.h>

// ---------------------------------------------------------------------------
// MaskDecoderLayer for MI455X (gfx1250): all GEMMs via v_wmma_f32_16x16x32_f16
// (f32 accumulate), attention softmax/LN statistics in f32.
// B=32, T=8, N=4096, D=256, H=2, DH=128.
// ---------------------------------------------------------------------------

typedef __attribute__((ext_vector_type(16))) _Float16 v16h;
typedef __attribute__((ext_vector_type(8)))  _Float16 v8h;
typedef __attribute__((ext_vector_type(8)))  float    v8f;

#define DEV __device__ __forceinline__

static constexpr int   Bz   = 32;
static constexpr int   Tz   = 8;
static constexpr int   Nimg = 4096;
static constexpr int   Dz   = 256;
static constexpr int   Hh   = 2;
static constexpr int   DH   = 128;
static constexpr float RSCALE = 0.08838834764831845f;  // 1/sqrt(128)

DEV v8f vzero8() { v8f z = {0.f,0.f,0.f,0.f,0.f,0.f,0.f,0.f}; return z; }

DEV v8f wmma_f16(v16h a, v16h b, v8f c) {
  return __builtin_amdgcn_wmma_f32_16x16x32_f16(false, a, false, b, (short)0, c,
                                                false, false);
}

// A-fragment loader (16x32 f16 tile, row-major source).
// Lane L: row = row_base + L%16; halves 0..7 = K (L/16)*8+0..7,
// halves 8..15 = K 16+(L/16)*8+0..7  (per CDNA5 ISA 16-bit A layout).
DEV v16h load_frag_A(const _Float16* A, int ld, int row_base, int kbase) {
  int lane = threadIdx.x & 31;
  int r  = row_base + (lane & 15);
  int hf = lane >> 4;
  const _Float16* p = A + (size_t)r * ld + kbase + hf * 8;
  union { v16h v; v8h h[2]; } u;
  u.h[0] = *(const v8h*)p;
  u.h[1] = *(const v8h*)(p + 16);
  return u.v;
}

// ---------------------------------------------------------------------------
// Generic WMMA GEMM: C[M,N] = A[M,K] * Bp + bias, A row-major f16,
// Bp pre-packed in fragment layout: ((nt*(K/32)+kt)*32 + lane)*16 halves.
// Wave tile 16(M) x 64(N), 4 accumulators; block = 4 waves = 64x64 tile.
// act: 0 = none, 1 = exact GELU.
// ---------------------------------------------------------------------------
__global__ __launch_bounds__(128)
void wmma_gemm_kernel(const _Float16* __restrict__ A,
                      const _Float16* __restrict__ Bp,
                      const float* __restrict__ bias,
                      float* __restrict__ Cf, _Float16* __restrict__ Ch,
                      int M, int N, int K, int act) {
  int wave = threadIdx.x >> 5;
  int lane = threadIdx.x & 31;
  int m_base = (blockIdx.y * 4 + wave) * 16;
  int n_base = blockIdx.x * 64;
  int KT = K >> 5;

  v8f acc0 = vzero8(), acc1 = vzero8(), acc2 = vzero8(), acc3 = vzero8();
  int nt0 = n_base >> 4;
  for (int kt = 0; kt < KT; ++kt) {
    v16h a = load_frag_A(A, K, m_base, kt * 32);
    const _Float16* bbase = Bp + ((size_t)kt * 32 + lane) * 16;
    size_t tstride = (size_t)KT * 32 * 16;
    v16h b0 = *(const v16h*)(bbase + (size_t)(nt0 + 0) * tstride);
    v16h b1 = *(const v16h*)(bbase + (size_t)(nt0 + 1) * tstride);
    v16h b2 = *(const v16h*)(bbase + (size_t)(nt0 + 2) * tstride);
    v16h b3 = *(const v16h*)(bbase + (size_t)(nt0 + 3) * tstride);
    acc0 = wmma_f16(a, b0, acc0);
    acc1 = wmma_f16(a, b1, acc1);
    acc2 = wmma_f16(a, b2, acc2);
    acc3 = wmma_f16(a, b3, acc3);
  }

  int c  = lane & 15;
  int hf = lane >> 4;
  v8f accs[4] = {acc0, acc1, acc2, acc3};
#pragma unroll
  for (int i = 0; i < 4; ++i) {
#pragma unroll
    for (int v = 0; v < 8; ++v) {
      int row = m_base + v + 8 * hf;
      int col = n_base + i * 16 + c;
      float val = accs[i][v] + bias[col];
      if (act == 1) val = 0.5f * val * (1.0f + erff(val * 0.70710678118f));
      size_t idx = (size_t)row * N + col;
      if (Cf) Cf[idx] = val;
      if (Ch) Ch[idx] = (_Float16)val;
    }
  }
}

// ---------------------------------------------------------------------------
// Weight packing into B-fragment layout (K=256 fixed, KT=8).
// Head-type weights W[h][d][e] (2x256x128) -> 16 column tiles at nt_off.
// ---------------------------------------------------------------------------
__global__ __launch_bounds__(256)
void pack_b_heads_kernel(const float* __restrict__ W, _Float16* __restrict__ Bp,
                         int nt_off) {
  int i = blockIdx.x * 256 + threadIdx.x;      // 16*8*32*16 = 65536
  int id16 = i & 15;
  int lane = (i >> 4) & 31;
  int kt   = (i >> 9) & 7;
  int ntl  = i >> 12;
  int c  = lane & 15;
  int hf = lane >> 4;
  int k  = (id16 < 8) ? (kt * 32 + hf * 8 + id16)
                      : (kt * 32 + 16 + hf * 8 + (id16 - 8));
  int col = ntl * 16 + c;
  int h = col >> 7, e = col & 127;
  float val = W[(size_t)h * Dz * DH + (size_t)k * DH + e];
  Bp[(((size_t)(nt_off + ntl) * 8 + kt) * 32 + lane) * 16 + id16] = (_Float16)val;
}

// Plain [256][256] weights (Wo, W1, W2), K-major rows.
__global__ __launch_bounds__(256)
void pack_b_plain_kernel(const float* __restrict__ W, _Float16* __restrict__ Bp) {
  int i = blockIdx.x * 256 + threadIdx.x;      // 65536
  int id16 = i & 15;
  int lane = (i >> 4) & 31;
  int kt   = (i >> 9) & 7;
  int ntl  = i >> 12;
  int c  = lane & 15;
  int hf = lane >> 4;
  int k  = (id16 < 8) ? (kt * 32 + hf * 8 + id16)
                      : (kt * 32 + 16 + hf * 8 + (id16 - 8));
  int col = ntl * 16 + c;
  Bp[(((size_t)ntl * 8 + kt) * 32 + lane) * 16 + id16] = (_Float16)W[(size_t)k * Dz + col];
}

__global__ __launch_bounds__(256)
void copy_f32_kernel(const float* __restrict__ src, float* __restrict__ dst, int n) {
  int i = blockIdx.x * 256 + threadIdx.x;
  if (i < n) dst[i] = src[i];
}

__global__ __launch_bounds__(256)
void f32_to_f16_kernel(const float* __restrict__ src, _Float16* __restrict__ dst, int n) {
  int i = blockIdx.x * 256 + threadIdx.x;
  if (i < n) dst[i] = (_Float16)src[i];
}

// img_pos = img_embed + positional encoding, emitted as f16.
__global__ __launch_bounds__(256)
void posenc_kernel(const float* __restrict__ img, _Float16* __restrict__ outh) {
  int i = blockIdx.x * 256 + threadIdx.x;        // 32*4096*256 total
  int d = i & 255;
  int n = (i >> 8) & (Nimg - 1);
  float ang = (float)n * expf(-(2.0f * (float)d / 256.0f) * 9.210340371976184f);
  float pe = (d & 1) ? cosf(ang) : sinf(ang);
  outh[i] = (_Float16)(img[i] + pe);
}

// ---------------------------------------------------------------------------
// Token self-attention core (tiny: per (b,h) 8x8 scores).
// qkv layout: [row=b*8+t][m*256 + h*128 + e], m: 0=q 1=k 2=v.
// ---------------------------------------------------------------------------
__global__ __launch_bounds__(128)
void sa_attn_kernel(const float* __restrict__ qkv, _Float16* __restrict__ outh) {
  int b = blockIdx.x >> 1, h = blockIdx.x & 1;
  __shared__ float ls[8][8];
  __shared__ float lp[8][8];
  int tid = threadIdx.x;
  if (tid < 64) {
    int qi = tid >> 3, ki = tid & 7;
    const float* q = qkv + (size_t)(b * 8 + qi) * 768 + h * 128;
    const float* k = qkv + (size_t)(b * 8 + ki) * 768 + 256 + h * 128;
    float s = 0.f;
    for (int e = 0; e < 128; ++e) s += q[e] * k[e];
    ls[qi][ki] = s * RSCALE;
  }
  __syncthreads();
  if (tid < 8) {
    float mx = -1e30f;
    for (int k2 = 0; k2 < 8; ++k2) mx = fmaxf(mx, ls[tid][k2]);
    float ex[8], sum = 0.f;
    for (int k2 = 0; k2 < 8; ++k2) { ex[k2] = __expf(ls[tid][k2] - mx); sum += ex[k2]; }
    for (int k2 = 0; k2 < 8; ++k2) lp[tid][k2] = ex[k2] / sum;
  }
  __syncthreads();
  int e = tid;  // 0..127
  for (int qi = 0; qi < 8; ++qi) {
    float acc = 0.f;
    for (int ki = 0; ki < 8; ++ki)
      acc += lp[qi][ki] * qkv[(size_t)(b * 8 + ki) * 768 + 512 + h * 128 + e];
    outh[(size_t)(b * 8 + qi) * 256 + h * 128 + e] = (_Float16)acc;
  }
}

// Residual + LayerNorm; writes f32 and/or f16.
__global__ __launch_bounds__(256)
void resid_ln_kernel(const float* __restrict__ xres, const float* __restrict__ y,
                     const float* __restrict__ g, const float* __restrict__ bta,
                     float* __restrict__ of, _Float16* __restrict__ oh) {
  __shared__ float red[256];
  int row = blockIdx.x, c = threadIdx.x;
  float v = xres[(size_t)row * 256 + c] + y[(size_t)row * 256 + c];
  red[c] = v; __syncthreads();
  for (int s = 128; s > 0; s >>= 1) { if (c < s) red[c] += red[c + s]; __syncthreads(); }
  float mu = red[0] * (1.0f / 256.0f);
  __syncthreads();
  float d = v - mu;
  red[c] = d * d; __syncthreads();
  for (int s = 128; s > 0; s >>= 1) { if (c < s) red[c] += red[c + s]; __syncthreads(); }
  float var = red[0] * (1.0f / 256.0f);
  float o = d * rsqrtf(var + 1e-5f) * g[c] + bta[c];
  if (of) of[(size_t)row * 256 + c] = o;
  if (oh) oh[(size_t)row * 256 + c] = (_Float16)o;
}

// Zero-pad 8 token queries to 16 rows per (b,h): qpad[bh][16][128].
__global__ __launch_bounds__(256)
void qpad_kernel(const _Float16* __restrict__ qh, _Float16* __restrict__ qpad) {
  int i = blockIdx.x * 256 + threadIdx.x;        // 64*16*128 = 131072
  int e = i & 127, r = (i >> 7) & 15, bh = i >> 11;
  int b = bh >> 1, h = bh & 1;
  _Float16 v = (_Float16)0;
  if (r < 8) v = qh[(size_t)(b * 8 + r) * 256 + h * 128 + e];
  qpad[i] = v;
}

// ---------------------------------------------------------------------------
// ca1 scores: S[bh][qi][j] = Qpad(16x128) x K^T per 16-key tile. One wave/tile.
// K rows come straight from the packed projection buffer X:
//   X[row = b*4096+n][col], col: [0,256)=ca1.K  [256,512)=ca1.V  [512,768)=ca2.Q
// ---------------------------------------------------------------------------
__global__ __launch_bounds__(128)
void ca1_scores_kernel(const _Float16* __restrict__ qpad,
                       const _Float16* __restrict__ X, float* __restrict__ S) {
  int wid  = blockIdx.x * 4 + (threadIdx.x >> 5);
  int lane = threadIdx.x & 31;
  int bh = wid >> 8, jt = wid & 255;
  int b = bh >> 1, h = bh & 1;
  int c = lane & 15, hf = lane >> 4;

  v8f acc = vzero8();
  const _Float16* qb = qpad + (size_t)bh * 16 * 128;
  const _Float16* kb = X + (size_t)b * Nimg * 768 + h * 128;  // key rows, ld=768
  for (int kt = 0; kt < 4; ++kt) {
    v16h a = load_frag_A(qb, 128, 0, kt * 32);
    const _Float16* kp = kb + (size_t)(jt * 16 + c) * 768 + kt * 32 + hf * 8;
    union { v16h v; v8h h2[2]; } u;
    u.h2[0] = *(const v8h*)kp;
    u.h2[1] = *(const v8h*)(kp + 16);
    acc = wmma_f16(a, u.v, acc);
  }
  if (hf == 0) {  // rows 0..7 are the real queries
#pragma unroll
    for (int v = 0; v < 8; ++v)
      S[((size_t)bh * 8 + v) * Nimg + jt * 16 + c] = acc[v];
  }
}

// Row softmax over 4096 keys, f16 output probabilities.
__global__ __launch_bounds__(256)
void softmax_rows_kernel(const float* __restrict__ S, _Float16* __restrict__ P) {
  __shared__ float red[256];
  size_t row = blockIdx.x;
  const float* s = S + row * Nimg;
  _Float16* p = P + row * Nimg;
  int t = threadIdx.x;
  float mx = -1e30f;
  for (int j = t; j < Nimg; j += 256) mx = fmaxf(mx, s[j] * RSCALE);
  red[t] = mx; __syncthreads();
  for (int st = 128; st > 0; st >>= 1) { if (t < st) red[t] = fmaxf(red[t], red[t + st]); __syncthreads(); }
  mx = red[0]; __syncthreads();
  float sum = 0.f;
  for (int j = t; j < Nimg; j += 256) sum += __expf(s[j] * RSCALE - mx);
  red[t] = sum; __syncthreads();
  for (int st = 128; st > 0; st >>= 1) { if (t < st) red[t] += red[t + st]; __syncthreads(); }
  float inv = 1.0f / red[0];
  for (int j = t; j < Nimg; j += 256)
    p[j] = (_Float16)(__expf(s[j] * RSCALE - mx) * inv);
}

// ca1 AV: out[b,qi,h*128+e] = sum_j P * V (V at X col offset 256). One block/(b,h).
__global__ __launch_bounds__(128)
void ca1_av_kernel(const _Float16* __restrict__ P, const _Float16* __restrict__ X,
                   _Float16* __restrict__ outh) {
  int bh = blockIdx.x, b = bh >> 1, h = bh & 1;
  __shared__ float lp[8][128];
  int tid = threadIdx.x;  // = e
  float acc[8] = {0.f,0.f,0.f,0.f,0.f,0.f,0.f,0.f};
  const _Float16* vb = X + (size_t)b * Nimg * 768 + 256 + h * 128 + tid;
  for (int jb = 0; jb < Nimg; jb += 128) {
    __syncthreads();
    for (int qi = 0; qi < 8; ++qi)
      lp[qi][tid] = (float)P[((size_t)bh * 8 + qi) * Nimg + jb + tid];
    __syncthreads();
    for (int jj = 0; jj < 128; ++jj) {
      float v = (float)vb[(size_t)(jb + jj) * 768];
#pragma unroll
      for (int qi = 0; qi < 8; ++qi) acc[qi] += lp[qi][jj] * v;
    }
  }
  for (int qi = 0; qi < 8; ++qi)
    outh[(size_t)(b * 8 + qi) * 256 + h * 128 + tid] = (_Float16)acc[qi];
}

// Zero-pad ca2 token keys: kpad[bh][16][128] (rows 8..15 = 0).
__global__ __launch_bounds__(256)
void kpad_kernel(const _Float16* __restrict__ kvh, _Float16* __restrict__ kpad) {
  int i = blockIdx.x * 256 + threadIdx.x;        // 131072
  int e = i & 127, r = (i >> 7) & 15, bh = i >> 11;
  int b = bh >> 1, h = bh & 1;
  _Float16 v = (_Float16)0;
  if (r < 8) v = kvh[(size_t)(b * 8 + r) * 512 + h * 128 + e];
  kpad[i] = v;
}

// Pack ca2 V into B-fragment layout (K=32, 8 valid + pad) per (b,h):
// vpk[((bh*8 + nt)*32 + lane)*16 + idx].
__global__ __launch_bounds__(256)
void vpack_kernel(const _Float16* __restrict__ kvh, _Float16* __restrict__ vpk) {
  int i = blockIdx.x * 256 + threadIdx.x;        // 64*8*32*16 = 262144
  int id16 = i & 15;
  int lane = (i >> 4) & 31;
  int nt   = (i >> 9) & 7;
  int bh   = i >> 12;
  int b = bh >> 1, h = bh & 1;
  int c  = lane & 15;
  int hf = lane >> 4;
  int k  = (id16 < 8) ? (hf * 8 + id16) : (16 + hf * 8 + (id16 - 8));
  int e  = nt * 16 + c;
  _Float16 v = (_Float16)0;
  if (k < 8) v = kvh[(size_t)(b * 8 + k) * 512 + 256 + h * 128 + e];
  vpk[i] = v;
}

// ---------------------------------------------------------------------------
// ca2 attention, fully fused per 16-img-row tile: WMMA scores vs 8 padded
// token keys, in-register cross-lane softmax (width-16 shuffle groups match
// the D-matrix layout), LDS reshape of P into an A-fragment, 8 WMMAs of P x V.
// ---------------------------------------------------------------------------
__global__ __launch_bounds__(128)
void ca2_attn_kernel(const _Float16* __restrict__ X,
                     const _Float16* __restrict__ kpad,
                     const _Float16* __restrict__ vpk,
                     _Float16* __restrict__ o2) {
  __shared__ __align__(16) _Float16 lP[4][16 * 32];
  int wslot = threadIdx.x >> 5;
  int lane  = threadIdx.x & 31;
  int wid = blockIdx.x * 4 + wslot;
  int bh = wid >> 8, nt = wid & 255;
  int b = bh >> 1, h = bh & 1;
  int c = lane & 15, hf = lane >> 4;

  // scores: Q(img rows) x Kpad^T, K-dim = 128
  v8f sacc = vzero8();
  const _Float16* qb = X + (size_t)b * Nimg * 768 + 512 + h * 128;  // ld=768
  const _Float16* kb = kpad + (size_t)bh * 16 * 128;
  for (int kt = 0; kt < 4; ++kt) {
    v16h a = load_frag_A(qb, 768, nt * 16, kt * 32);
    const _Float16* kp = kb + (size_t)c * 128 + kt * 32 + hf * 8;
    union { v16h v; v8h h2[2]; } u;
    u.h2[0] = *(const v8h*)kp;
    u.h2[1] = *(const v8h*)(kp + 16);
    sacc = wmma_f16(a, u.v, sacc);
  }

  // softmax over the 8 valid token keys (columns), per output row.
  float p[8];
#pragma unroll
  for (int v = 0; v < 8; ++v) {
    float s = (c < 8) ? sacc[v] * RSCALE : -1e30f;
    float m = s;
    for (int off = 1; off < 16; off <<= 1) m = fmaxf(m, __shfl_xor(m, off, 16));
    float e = (c < 8) ? __expf(s - m) : 0.f;
    float sum = e;
    for (int off = 1; off < 16; off <<= 1) sum += __shfl_xor(sum, off, 16);
    p[v] = e / sum;
  }

  // Reshape P (D layout) -> A-fragment source in LDS, K padded to 32.
  _Float16* lp = &lP[wslot][0];
#pragma unroll
  for (int v = 0; v < 8; ++v) lp[(v + 8 * hf) * 32 + c] = (_Float16)p[v];
#pragma unroll
  for (int z = 0; z < 8; ++z) lp[(lane & 15) * 32 + 16 + hf * 8 + z] = (_Float16)0;
  __syncthreads();

  union { v16h v; v8h h2[2]; } ua;
  const _Float16* lr = lp + (size_t)(lane & 15) * 32;
  ua.h2[0] = *(const v8h*)(lr + hf * 8);
  ua.h2[1] = *(const v8h*)(lr + 16 + hf * 8);
  v16h pa = ua.v;

  // out[16 x 128] = P x V  (8 column tiles, single K=32 step)
  const _Float16* vb = vpk + (size_t)bh * 8 * 32 * 16;
#pragma unroll
  for (int i = 0; i < 8; ++i) {
    v16h vbfrag = *(const v16h*)(vb + (((size_t)i * 32) + lane) * 16);
    v8f oacc = wmma_f16(pa, vbfrag, vzero8());
#pragma unroll
    for (int v = 0; v < 8; ++v) {
      int row = nt * 16 + v + 8 * hf;
      o2[((size_t)(b * Nimg + row)) * 256 + h * 128 + i * 16 + c] = (_Float16)oacc[v];
    }
  }
}

// ---------------------------------------------------------------------------
// Host orchestration
// ---------------------------------------------------------------------------
static inline dim3 gemm_grid(int M, int N) { return dim3(N / 64, M / 64); }

extern "C" void kernel_launch(void* const* d_in, const int* in_sizes, int n_in,
                              void* d_out, int out_size, void* d_ws, size_t ws_size,
                              hipStream_t stream) {
  (void)in_sizes; (void)n_in; (void)out_size; (void)ws_size;
  const float* tokens = (const float*)d_in[0];
  const float* img    = (const float*)d_in[1];
  // sa
  const float* saWk = (const float*)d_in[2];  const float* sabk = (const float*)d_in[3];
  const float* saWq = (const float*)d_in[4];  const float* sabq = (const float*)d_in[5];
  const float* saWv = (const float*)d_in[6];  const float* sabv = (const float*)d_in[7];
  const float* saWo = (const float*)d_in[8];  const float* sabo = (const float*)d_in[9];
  // ca1
  const float* c1Wk = (const float*)d_in[10]; const float* c1bk = (const float*)d_in[11];
  const float* c1Wq = (const float*)d_in[12]; const float* c1bq = (const float*)d_in[13];
  const float* c1Wv = (const float*)d_in[14]; const float* c1bv = (const float*)d_in[15];
  const float* c1Wo = (const float*)d_in[16]; const float* c1bo = (const float*)d_in[17];
  // ca2
  const float* c2Wk = (const float*)d_in[18]; const float* c2bk = (const float*)d_in[19];
  const float* c2Wq = (const float*)d_in[20]; const float* c2bq = (const float*)d_in[21];
  const float* c2Wv = (const float*)d_in[22]; const float* c2bv = (const float*)d_in[23];
  const float* c2Wo = (const float*)d_in[24]; const float* c2bo = (const float*)d_in[25];
  const float* W1 = (const float*)d_in[26]; const float* b1 = (const float*)d_in[27];
  const float* W2 = (const float*)d_in[28]; const float* b2 = (const float*)d_in[29];
  const float* ln1g = (const float*)d_in[30]; const float* ln1b = (const float*)d_in[31];
  const float* ln2g = (const float*)d_in[32]; const float* ln2b = (const float*)d_in[33];
  const float* ln3g = (const float*)d_in[34]; const float* ln3b = (const float*)d_in[35];

  float* out_t   = (float*)d_out;          // [256,256]
  float* out_i2t = (float*)d_out + 65536;  // [131072,256]

  char* ws = (char*)d_ws;
  size_t off = 0;
#define WS_ALLOC(var, type, count) \
  type* var = (type*)(ws + off);   \
  off += (((size_t)(count) * sizeof(type)) + 255) & ~(size_t)255;

  const int Mtok = Bz * Tz;       // 256
  const int Mimg = Bz * Nimg;     // 131072

  WS_ALLOC(tokens_h, _Float16, Mtok * 256);
  WS_ALLOC(Bp_sa,    _Float16, 256 * 768);  WS_ALLOC(bias_sa,  float, 768);
  WS_ALLOC(Bp_q1,    _Float16, 256 * 256);  WS_ALLOC(bias_q1,  float, 256);
  WS_ALLOC(Bp_big1,  _Float16, 256 * 768);  WS_ALLOC(bias_big1, float, 768);
  WS_ALLOC(Bp_kv2,   _Float16, 256 * 512);  WS_ALLOC(bias_kv2, float, 512);
  WS_ALLOC(Bp_wos,   _Float16, 256 * 256);  WS_ALLOC(bias_wos, float, 256);
  WS_ALLOC(Bp_wo1,   _Float16, 256 * 256);  WS_ALLOC(bias_wo1, float, 256);
  WS_ALLOC(Bp_wo2,   _Float16, 256 * 256);  WS_ALLOC(bias_wo2, float, 256);
  WS_ALLOC(Bp_w1,    _Float16, 256 * 256);  WS_ALLOC(bias_w1,  float, 256);
  WS_ALLOC(Bp_w2,    _Float16, 256 * 256);  WS_ALLOC(bias_w2,  float, 256);
  WS_ALLOC(qkv_sa,   float,    Mtok * 768);
  WS_ALLOC(attn_sa_h,_Float16, Mtok * 256);
  WS_ALLOC(tmp1,     float,    Mtok * 256);
  WS_ALLOC(t1,       float,    Mtok * 256);  WS_ALLOC(t1_h, _Float16, Mtok * 256);
  WS_ALLOC(t2,       float,    Mtok * 256);  WS_ALLOC(t2_h, _Float16, Mtok * 256);
  WS_ALLOC(t3_h,     _Float16, Mtok * 256);
  WS_ALLOC(qc1_h,    _Float16, Mtok * 256);
  WS_ALLOC(qpad,     _Float16, 64 * 16 * 128);
  WS_ALLOC(t2i_h,    _Float16, Mtok * 256);
  WS_ALLOC(hgelu_h,  _Float16, Mtok * 256);
  WS_ALLOC(kvh,      _Float16, Mtok * 512);
  WS_ALLOC(kpad,     _Float16, 64 * 16 * 128);
  WS_ALLOC(vpk,      _Float16, 64 * 8 * 32 * 16);
  WS_ALLOC(Sbuf,     float,    512 * Nimg);
  WS_ALLOC(Pbuf,     _Float16, 512 * Nimg);
  WS_ALLOC(img_pos_h,_Float16, (size_t)Mimg * 256);
  WS_ALLOC(Xbuf,     _Float16, (size_t)Mimg * 768);
  WS_ALLOC(o2buf,    _Float16, (size_t)Mimg * 256);
#undef WS_ALLOC

  // ---- weight packing (small, once per launch) ----
  dim3 pblk(256), pgrd(256);  // 65536 threads
  pack_b_heads_kernel<<<pgrd, pblk, 0, stream>>>(saWq, Bp_sa, 0);
  pack_b_heads_kernel<<<pgrd, pblk, 0, stream>>>(saWk, Bp_sa, 16);
  pack_b_heads_kernel<<<pgrd, pblk, 0, stream>>>(saWv, Bp_sa, 32);
  copy_f32_kernel<<<1, 256, 0, stream>>>(sabq, bias_sa + 0,   256);
  copy_f32_kernel<<<1, 256, 0, stream>>>(sabk, bias_sa + 256, 256);
  copy_f32_kernel<<<1, 256, 0, stream>>>(sabv, bias_sa + 512, 256);
  pack_b_heads_kernel<<<pgrd, pblk, 0, stream>>>(c1Wq, Bp_q1, 0);
  copy_f32_kernel<<<1, 256, 0, stream>>>(c1bq, bias_q1, 256);
  pack_b_heads_kernel<<<pgrd, pblk, 0, stream>>>(c1Wk, Bp_big1, 0);
  pack_b_heads_kernel<<<pgrd, pblk, 0, stream>>>(c1Wv, Bp_big1, 16);
  pack_b_heads_kernel<<<pgrd, pblk, 0, stream>>>(c2Wq, Bp_big1, 32);
  copy_f32_kernel<<<1, 256, 0, stream>>>(c1bk, bias_big1 + 0,   256);
  copy_f32_kernel<<<1, 256, 0, stream>>>(c1bv, bias_big1 + 256, 256);
  copy_f32_kernel<<<1, 256, 0, stream>>>(c2bq, bias_big1 + 512, 256);
  pack_b_heads_kernel<<<pgrd, pblk, 0, stream>>>(c2Wk, Bp_kv2, 0);
  pack_b_heads_kernel<<<pgrd, pblk, 0, stream>>>(c2Wv, Bp_kv2, 16);
  copy_f32_kernel<<<1, 256, 0, stream>>>(c2bk, bias_kv2 + 0,   256);
  copy_f32_kernel<<<1, 256, 0, stream>>>(c2bv, bias_kv2 + 256, 256);
  pack_b_plain_kernel<<<pgrd, pblk, 0, stream>>>(saWo, Bp_wos);
  copy_f32_kernel<<<1, 256, 0, stream>>>(sabo, bias_wos, 256);
  pack_b_plain_kernel<<<pgrd, pblk, 0, stream>>>(c1Wo, Bp_wo1);
  copy_f32_kernel<<<1, 256, 0, stream>>>(c1bo, bias_wo1, 256);
  pack_b_plain_kernel<<<pgrd, pblk, 0, stream>>>(c2Wo, Bp_wo2);
  copy_f32_kernel<<<1, 256, 0, stream>>>(c2bo, bias_wo2, 256);
  pack_b_plain_kernel<<<pgrd, pblk, 0, stream>>>(W1, Bp_w1);
  copy_f32_kernel<<<1, 256, 0, stream>>>(b1, bias_w1, 256);
  pack_b_plain_kernel<<<pgrd, pblk, 0, stream>>>(W2, Bp_w2);
  copy_f32_kernel<<<1, 256, 0, stream>>>(b2, bias_w2, 256);

  // ---- img-side heavy path ----
  posenc_kernel<<<Mimg * 256 / 256, 256, 0, stream>>>(img, img_pos_h);
  wmma_gemm_kernel<<<gemm_grid(Mimg, 768), 128, 0, stream>>>(
      img_pos_h, Bp_big1, bias_big1, nullptr, Xbuf, Mimg, 768, 256, 0);

  // ---- token self-attention + LN1 ----
  f32_to_f16_kernel<<<Mtok * 256 / 256, 256, 0, stream>>>(tokens, tokens_h, Mtok * 256);
  wmma_gemm_kernel<<<gemm_grid(Mtok, 768), 128, 0, stream>>>(
      tokens_h, Bp_sa, bias_sa, qkv_sa, nullptr, Mtok, 768, 256, 0);
  sa_attn_kernel<<<64, 128, 0, stream>>>(qkv_sa, attn_sa_h);
  wmma_gemm_kernel<<<gemm_grid(Mtok, 256), 128, 0, stream>>>(
      attn_sa_h, Bp_wos, bias_wos, tmp1, nullptr, Mtok, 256, 256, 0);
  resid_ln_kernel<<<Mtok, 256, 0, stream>>>(tokens, tmp1, ln1g, ln1b, t1, t1_h);

  // ---- ca1: tokens attend to image ----
  wmma_gemm_kernel<<<gemm_grid(Mtok, 256), 128, 0, stream>>>(
      t1_h, Bp_q1, bias_q1, nullptr, qc1_h, Mtok, 256, 256, 0);
  qpad_kernel<<<131072 / 256, 256, 0, stream>>>(qc1_h, qpad);
  ca1_scores_kernel<<<64 * 256 / 4, 128, 0, stream>>>(qpad, Xbuf, Sbuf);
  softmax_rows_kernel<<<512, 256, 0, stream>>>(Sbuf, Pbuf);
  ca1_av_kernel<<<64, 128, 0, stream>>>(Pbuf, Xbuf, t2i_h);
  wmma_gemm_kernel<<<gemm_grid(Mtok, 256), 128, 0, stream>>>(
      t2i_h, Bp_wo1, bias_wo1, tmp1, nullptr, Mtok, 256, 256, 0);
  resid_ln_kernel<<<Mtok, 256, 0, stream>>>(t1, tmp1, ln2g, ln2b, t2, t2_h);

  // ---- MLP + LN3 (t output) ----
  wmma_gemm_kernel<<<gemm_grid(Mtok, 256), 128, 0, stream>>>(
      t2_h, Bp_w1, bias_w1, nullptr, hgelu_h, Mtok, 256, 256, 1 /*gelu*/);
  wmma_gemm_kernel<<<gemm_grid(Mtok, 256), 128, 0, stream>>>(
      hgelu_h, Bp_w2, bias_w2, tmp1, nullptr, Mtok, 256, 256, 0);
  resid_ln_kernel<<<Mtok, 256, 0, stream>>>(t2, tmp1, ln3g, ln3b, out_t, t3_h);

  // ---- ca2: image attends to tokens ----
  wmma_gemm_kernel<<<gemm_grid(Mtok, 512), 128, 0, stream>>>(
      t3_h, Bp_kv2, bias_kv2, nullptr, kvh, Mtok, 512, 256, 0);
  kpad_kernel<<<131072 / 256, 256, 0, stream>>>(kvh, kpad);
  vpack_kernel<<<262144 / 256, 256, 0, stream>>>(kvh, vpk);
  ca2_attn_kernel<<<64 * 256 / 4, 128, 0, stream>>>(Xbuf, kpad, vpk, o2buf);
  wmma_gemm_kernel<<<gemm_grid(Mimg, 256), 128, 0, stream>>>(
      o2buf, Bp_wo2, bias_wo2, out_i2t, nullptr, Mimg, 256, 256, 0);
}
